// SelfAttention_86423331930529
// MI455X (gfx1250) — matrix operand order repaired and checked by reading, not compiled, not run on hardware
//
#include <hip/hip_runtime.h>

typedef __bf16 bf16;
typedef __attribute__((ext_vector_type(16))) __bf16 v16bf;
typedef __attribute__((ext_vector_type(8)))  __bf16 v8bf;
typedef __attribute__((ext_vector_type(8)))  float  v8f;

#define B_   2
#define S_   2048
#define H_   1024
#define NH_  16
#define HD_  64
#define MASK_VALUE 10000.0f

// ---------------------------------------------------------------------------
// WMMA helper: D = A(16x32 bf16) * B(32x16 bf16) + C(16x16 f32)
// ---------------------------------------------------------------------------
__device__ __forceinline__ v8f wmma_bf16(v16bf a, v16bf b, v8f c) {
  return __builtin_amdgcn_wmma_f32_16x16x32_bf16(
      /*neg_a=*/false, a, /*neg_b=*/false, b,
      /*c_mod=*/(short)0, c, /*reuse_a=*/false, /*reuse_b=*/false);
}

// A fragment: 16(M) x 32(K) bf16. Source is row-major [m][k] with `stride`
// halves per row. Per ISA: lanes 0-15 hold M=lane, K={0..7,16..23};
// lanes 16-31 hold M=lane-16, K={8..15,24..31}.  Two 16B loads per lane.
__device__ __forceinline__ v16bf load_a_frag(const bf16* base, int row0,
                                             int stride, int kOff, int lane) {
  int m     = lane & 15;
  int khalf = (lane >> 4) << 3;  // 0 or 8
  const bf16* p = base + (size_t)(row0 + m) * stride + kOff;
  union { v16bf v; v8bf h[2]; } u;
  u.h[0] = *(const v8bf*)(p + khalf);        // K = khalf + 0..7
  u.h[1] = *(const v8bf*)(p + 16 + khalf);   // K = 16 + khalf + 0..7
  return u.v;
}

// B fragment: 32(K) x 16(N) bf16. Source stored [n][k]-major with `stride`
// halves per n. Lanes 0-15: N=lane, K={0..7,16..23}; lanes 16-31: N=lane-16, K={8..15,24..31}.
__device__ __forceinline__ v16bf load_b_frag(const bf16* base, int col0,
                                             int stride, int kOff, int lane) {
  int n  = lane & 15;
  int kh = (lane >> 4) << 3;  // 0 or 8
  const bf16* p = base + (size_t)(col0 + n) * stride + kOff + kh;
  union { v16bf v; v8bf h[2]; } u;
  u.h[0] = *(const v8bf*)(p);
  u.h[1] = *(const v8bf*)(p + 16);
  return u.v;
}

// ---------------------------------------------------------------------------
// Tiled GEMM:  C[M,N] = A[M,K] * B[K,N] + bias[N]
//   MODE 0: A = fp32 x, output scattered to Q/K/V head-major bf16.
//   MODE 1: A = bf16 ctx, output fp32 row-major.
// Block tile 128x128, K-step 32, 256 threads (8 waves, 4x2 wave grid,
// each wave computes a 32x64 region = 2x4 WMMA tiles).
// ---------------------------------------------------------------------------
template <int MODE>
__global__ __launch_bounds__(256) void gemm_bf16_kernel(
    const float* __restrict__ Af, const bf16* __restrict__ Ab,
    const float* __restrict__ Bw, const float* __restrict__ bias,
    int M, int N, int K,
    bf16* __restrict__ Qo, bf16* __restrict__ Ko, bf16* __restrict__ Vo,
    float* __restrict__ Co) {
  __shared__ bf16 sA[128 * 32];  // [m][k]
  __shared__ bf16 sB[128 * 32];  // [n][k] (transposed at stage time)

  const int tid  = threadIdx.x;
  const int lane = tid & 31;
  const int w    = tid >> 5;
  const int wm   = (w & 3) * 32;   // wave M offset in block tile
  const int wn   = (w >> 2) * 64;  // wave N offset in block tile
  const int rowBase = blockIdx.y * 128;
  const int colBase = blockIdx.x * 128;

  v8f acc[2][4];
#pragma unroll
  for (int mt = 0; mt < 2; ++mt)
#pragma unroll
    for (int nt = 0; nt < 4; ++nt)
#pragma unroll
      for (int i = 0; i < 8; ++i) acc[mt][nt][i] = 0.f;

  for (int k0 = 0; k0 < K; k0 += 32) {
    // Stage A (128x32) and B (32x128 -> [n][k]) tiles as bf16.
#pragma unroll
    for (int i = 0; i < 16; ++i) {
      int idx = tid + i * 256;
      int ra = idx >> 5, ca = idx & 31;
      float va = (MODE == 0) ? Af[(size_t)(rowBase + ra) * K + k0 + ca]
                             : (float)Ab[(size_t)(rowBase + ra) * K + k0 + ca];
      sA[ra * 32 + ca] = (bf16)va;
      int cb = idx & 127, rb = idx >> 7;
      sB[cb * 32 + rb] = (bf16)Bw[(size_t)(k0 + rb) * N + colBase + cb];
    }
    __syncthreads();

    v16bf afr[2], bfr[4];
#pragma unroll
    for (int mt = 0; mt < 2; ++mt)
      afr[mt] = load_a_frag(sA, wm + mt * 16, 32, 0, lane);
#pragma unroll
    for (int nt = 0; nt < 4; ++nt)
      bfr[nt] = load_b_frag(sB, wn + nt * 16, 32, 0, lane);
#pragma unroll
    for (int mt = 0; mt < 2; ++mt)
#pragma unroll
      for (int nt = 0; nt < 4; ++nt)
        acc[mt][nt] = wmma_bf16(afr[mt], bfr[nt], acc[mt][nt]);
    __syncthreads();
  }

  // Epilogue. C layout: vgpr r, lane l -> row = r + 8*(l>=16), col = l%16.
  const int lcol = lane & 15;
  const int lrow = (lane >> 4) << 3;
#pragma unroll
  for (int mt = 0; mt < 2; ++mt) {
#pragma unroll
    for (int nt = 0; nt < 4; ++nt) {
      int col = colBase + wn + nt * 16 + lcol;
      float bv = bias[col];
#pragma unroll
      for (int r = 0; r < 8; ++r) {
        int row = rowBase + wm + mt * 16 + r + lrow;
        float v = acc[mt][nt][r] + bv;
        if (MODE == 0) {
          // col in [0,3072): segment(q/k/v) | head | dim
          int seg = col >> 10;
          int hh  = (col & 1023) >> 6;
          int dd  = col & 63;
          int bb  = row >> 11;        // / S_
          int ss  = row & (S_ - 1);
          bf16* dst = (seg == 0) ? Qo : (seg == 1) ? Ko : Vo;
          dst[(((size_t)bb * NH_ + hh) * S_ + ss) * HD_ + dd] = (bf16)v;
        } else {
          Co[(size_t)row * N + col] = v;
        }
      }
    }
  }
}

// ---------------------------------------------------------------------------
// Flash attention, causal. One block per (q-tile of 64 rows, head, batch).
// 4 waves; wave w owns q rows [qb + 16w, qb + 16w + 16).
// ---------------------------------------------------------------------------
__global__ __launch_bounds__(128) void attn_kernel(
    const bf16* __restrict__ Q, const bf16* __restrict__ K,
    const bf16* __restrict__ V, bf16* __restrict__ Ctx) {
  __shared__ bf16 sK[64 * 64];      // [key][d]  (B-frag layout for Q*K^T)
  __shared__ bf16 sV[64 * 64];      // [d][key]  (B-frag layout for P*V)
  __shared__ bf16 sP[4][16 * 64];   // per-wave P transpose buffer [m][key]

  const int tid  = threadIdx.x;
  const int lane = tid & 31;
  const int w    = tid >> 5;
  const int qb   = blockIdx.x * 64;
  const int head = blockIdx.y;
  const int bat  = blockIdx.z;

  const size_t hoff = ((size_t)bat * NH_ + head) * S_ * HD_;
  const bf16* Qh = Q + hoff;
  const bf16* Kh = K + hoff;
  const bf16* Vh = V + hoff;

  // Q fragments for this wave's 16 rows (two 32-wide d-chunks), from global.
  v16bf aQ[2];
#pragma unroll
  for (int c = 0; c < 2; ++c)
    aQ[c] = load_a_frag(Qh + (size_t)(qb + w * 16) * HD_, 0, HD_, c * 32, lane);

  const int lcol = lane & 15;
  const int lrow = (lane >> 4) << 3;  // 0 or 8

  float mrun[8], lsum[8];
  v8f o[4];
#pragma unroll
  for (int r = 0; r < 8; ++r) { mrun[r] = -1e30f; lsum[r] = 0.f; }
#pragma unroll
  for (int nt = 0; nt < 4; ++nt)
#pragma unroll
    for (int i = 0; i < 8; ++i) o[nt][i] = 0.f;

  const int nkb = (qb >> 6) + 1;  // causal: only k-blocks at or below q-tile
  for (int kb = 0; kb < nkb; ++kb) {
    // Stage K tile [key][d] and V tile transposed [d][key].
#pragma unroll
    for (int i = 0; i < 32; ++i) {
      int idx = tid + i * 128;
      int d = idx & 63, kk = idx >> 6;
      sK[kk * 64 + d] = Kh[(size_t)(kb * 64 + kk) * HD_ + d];
      sV[d * 64 + kk] = Vh[(size_t)(kb * 64 + kk) * HD_ + d];
    }
    __syncthreads();

    // Scores: S = Q (16x64) * K^T (64 x 64keys) -> 4 tiles of 16x16.
    v8f sc[4];
#pragma unroll
    for (int nt = 0; nt < 4; ++nt) {
      v8f t;
#pragma unroll
      for (int i = 0; i < 8; ++i) t[i] = 0.f;
      t = wmma_bf16(aQ[0], load_b_frag(sK, nt * 16, 64, 0, lane), t);
      t = wmma_bf16(aQ[1], load_b_frag(sK, nt * 16, 64, 32, lane), t);
      sc[nt] = t;
    }

    // Scale + causal mask + online softmax (rows live across 16-lane groups).
#pragma unroll
    for (int r = 0; r < 8; ++r) {
      int grow = qb + w * 16 + r + lrow;
      float mx = mrun[r];
#pragma unroll
      for (int nt = 0; nt < 4; ++nt) {
        int gcol = kb * 64 + nt * 16 + lcol;
        float val = (gcol <= grow) ? sc[nt][r] * 0.125f : -MASK_VALUE;
        sc[nt][r] = val;
        mx = fmaxf(mx, val);
      }
#pragma unroll
      for (int off = 8; off > 0; off >>= 1)
        mx = fmaxf(mx, __shfl_xor(mx, off, 16));

      float corr = __expf(mrun[r] - mx);
      mrun[r] = mx;
      lsum[r] *= corr;
#pragma unroll
      for (int nt = 0; nt < 4; ++nt) o[nt][r] *= corr;

      float ls = 0.f;
#pragma unroll
      for (int nt = 0; nt < 4; ++nt) {
        float p = __expf(sc[nt][r] - mx);
        sc[nt][r] = p;
        ls += p;
      }
#pragma unroll
      for (int off = 8; off > 0; off >>= 1) ls += __shfl_xor(ls, off, 16);
      lsum[r] += ls;
    }

    // P: C-layout -> A-layout via per-wave LDS transpose buffer.
#pragma unroll
    for (int r = 0; r < 8; ++r)
#pragma unroll
      for (int nt = 0; nt < 4; ++nt)
        sP[w][(r + lrow) * 64 + nt * 16 + lcol] = (bf16)sc[nt][r];
    __syncthreads();  // orders cross-lane sP writes; all sK reads done

    v16bf aP0 = load_a_frag(sP[w], 0, 64, 0, lane);
    v16bf aP1 = load_a_frag(sP[w], 0, 64, 32, lane);
#pragma unroll
    for (int nt = 0; nt < 4; ++nt) {
      o[nt] = wmma_bf16(aP0, load_b_frag(sV, nt * 16, 64, 0, lane), o[nt]);
      o[nt] = wmma_bf16(aP1, load_b_frag(sV, nt * 16, 64, 32, lane), o[nt]);
    }
    __syncthreads();  // sV reads done before next stage overwrites
  }

  // Normalize and write ctx in [b, s, h*64+d] layout (bf16) for out-proj.
#pragma unroll
  for (int r = 0; r < 8; ++r) {
    float inv = 1.f / lsum[r];
    int srow = qb + w * 16 + r + lrow;
#pragma unroll
    for (int nt = 0; nt < 4; ++nt) {
      int d = nt * 16 + lcol;
      Ctx[((size_t)(bat * S_ + srow)) * H_ + head * HD_ + d] =
          (bf16)(o[nt][r] * inv);
    }
  }
}

// ---------------------------------------------------------------------------
// Inputs (setup_inputs order): x, mask(unused), W_qkv, b_qkv, W_o, b_o.
// Workspace: Q | K | V | ctx, all bf16 head-major / row-major (32 MB).
// ---------------------------------------------------------------------------
extern "C" void kernel_launch(void* const* d_in, const int* in_sizes, int n_in,
                              void* d_out, int out_size, void* d_ws,
                              size_t ws_size, hipStream_t stream) {
  const float* x     = (const float*)d_in[0];
  const float* W_qkv = (const float*)d_in[2];
  const float* b_qkv = (const float*)d_in[3];
  const float* W_o   = (const float*)d_in[4];
  const float* b_o   = (const float*)d_in[5];
  float* out = (float*)d_out;

  const size_t perTensor = (size_t)B_ * NH_ * S_ * HD_;  // 4M elems
  bf16* Qb  = (bf16*)d_ws;
  bf16* Kb  = Qb + perTensor;
  bf16* Vb  = Kb + perTensor;
  bf16* Ctx = Vb + perTensor;

  const int M = B_ * S_;  // 4096

  dim3 g1(3 * H_ / 128, M / 128);  // 24 x 32
  gemm_bf16_kernel<0><<<g1, 256, 0, stream>>>(x, nullptr, W_qkv, b_qkv,
                                              M, 3 * H_, H_,
                                              Qb, Kb, Vb, nullptr);

  dim3 g2(S_ / 64, NH_, B_);  // 32 x 16 x 2
  attn_kernel<<<g2, 128, 0, stream>>>(Qb, Kb, Vb, Ctx);

  dim3 g3(H_ / 128, M / 128);  // 8 x 32
  gemm_bf16_kernel<1><<<g3, 256, 0, stream>>>(nullptr, Ctx, W_o, b_o,
                                              M, H_, H_,
                                              nullptr, nullptr, nullptr, out);
}
